// GraphTransformer_5995774345344
// MI455X (gfx1250) — matrix operand order repaired and checked by reading, not compiled, not run on hardware
//
#include <hip/hip_runtime.h>
#include <math.h>

// ---------------------------------------------------------------------------
// Graph TransformerConv (heads=1) for MI455X / gfx1250.
// Dense GEMMs via v_wmma_f32_16x16x32_bf16 with bf16 hi/lo error compensation
// (~fp32 accuracy, fp32 accumulate). Edge attention phase is bandwidth-bound
// and stays fp32 with float4 gathers + global f32 atomics.
// ---------------------------------------------------------------------------

typedef __attribute__((ext_vector_type(16))) __bf16 v16bf;
typedef __attribute__((ext_vector_type(8)))  __bf16 v8bf;
typedef __attribute__((ext_vector_type(8)))  float  v8f;

#define FD   256   // feature dim
#define NT_W 8     // 16-col output tiles per wave (wave covers 128 cols)

union BfFrag { v16bf v; v8bf h[2]; };

// ---------------- weight prep: transpose + bf16 hi/lo split -----------------
// Output layout: Wt[lw][n][k] = W[layer][k][n], lw = layer*4 + which
__global__ void prep_w_kernel(const float* __restrict__ Wq, const float* __restrict__ Wk,
                              const float* __restrict__ Wv, const float* __restrict__ Ws,
                              __bf16* __restrict__ Whi, __bf16* __restrict__ Wlo, int L)
{
    long long idx = (long long)blockIdx.x * 256 + threadIdx.x;
    long long total = (long long)L * 4 * FD * FD;
    if (idx >= total) return;
    int lw     = (int)(idx >> 16);
    int within = (int)(idx & 65535);
    int n  = within >> 8;
    int kk = within & 255;
    int layer = lw >> 2, which = lw & 3;
    const float* Wsrc = (which == 0) ? Wq : (which == 1) ? Wk : (which == 2) ? Wv : Ws;
    float val = Wsrc[(size_t)layer * FD * FD + (size_t)kk * FD + n];
    __bf16 h = (__bf16)val;
    Whi[idx] = h;
    Wlo[idx] = (__bf16)(val - (float)h);
}

// ---------------- input conversion: fp32 -> bf16 hi/lo ----------------------
__global__ void cvt_kernel(const float* __restrict__ in, __bf16* __restrict__ hi,
                           __bf16* __restrict__ lo, long long n)
{
    long long i = (long long)blockIdx.x * 256 + threadIdx.x;
    if (i >= n) return;
    float v = in[i];
    __bf16 h = (__bf16)v;
    hi[i] = h;
    lo[i] = (__bf16)(v - (float)h);
}

// ---------------- WMMA GEMM: out = Ahi/lo @ Wt^T + bias ---------------------
// A: [Nrows x 256] bf16 hi/lo, row-major. Wt: [256 x 256] k-major (transposed).
// Each wave: one 16-row M tile x 128 cols (blockIdx.y picks column half).
// Weights staged through LDS (16 KB) per 32-wide K step.
__global__ void __launch_bounds__(256)
gemm_bf16x3_kernel(const __bf16* __restrict__ Ahi, const __bf16* __restrict__ Alo,
                   const __bf16* __restrict__ Whi, const __bf16* __restrict__ Wlo,
                   const float* __restrict__ bias, float* __restrict__ out, int Nrows)
{
    __shared__ __attribute__((aligned(16))) __bf16 ldsW[2 * 128 * 32]; // [sel][col][k]
    const int tid  = threadIdx.x;
    const int wid  = tid >> 5;
    const int lane = tid & 31;
    const int hi   = lane >> 4;
    const int l16  = lane & 15;
    const int mt_cnt = (Nrows + 15) >> 4;
    const int mtile  = blockIdx.x * 8 + wid;
    const bool active = (mtile < mt_cnt);
    const int n0 = blockIdx.y * 128;
    int row = mtile * 16 + l16;
    if (row >= Nrows) row = Nrows - 1;

    v8f acc[NT_W] = {};

    for (int kb = 0; kb < 8; ++kb) {
        const int kbase = kb * 32;
        __syncthreads();
        // Cooperative stage of 128-col x 32-K weight chunk (hi + lo) into LDS.
        #pragma unroll
        for (int i = 0; i < 4; ++i) {
            int c   = tid + i * 256;        // 0..1023 chunks of 8 bf16
            int sel = c >> 9;               // 0 = hi, 1 = lo
            int rem = c & 511;
            int n   = rem >> 2;             // 0..127 local col
            int kk  = (rem & 3) * 8;        // 0,8,16,24
            const __bf16* src = (sel ? Wlo : Whi) + (size_t)(n0 + n) * FD + kbase + kk;
            *(uint4*)&ldsW[sel * 4096 + n * 32 + kk] = *(const uint4*)src;
        }
        __syncthreads();
        if (active) {
            // A fragment: lanes 0-15 hold K {0..7,16..23}, lanes 16-31 {8..15,24..31}
            BfFrag ahi, alo;
            const __bf16* arh = Ahi + (size_t)row * FD + kbase;
            const __bf16* arl = Alo + (size_t)row * FD + kbase;
            ahi.h[0] = *(const v8bf*)(arh + 8 * hi);
            ahi.h[1] = *(const v8bf*)(arh + 16 + 8 * hi);
            alo.h[0] = *(const v8bf*)(arl + 8 * hi);
            alo.h[1] = *(const v8bf*)(arl + 16 + 8 * hi);
            #pragma unroll
            for (int nt = 0; nt < NT_W; ++nt) {
                int cl = nt * 16 + l16;
                // B fragment: lane holds one column, K {8*hi..8*hi+7, 16+8*hi..16+8*hi+7}
                BfFrag bhi, blo;
                const __bf16* bh = &ldsW[0 * 4096 + cl * 32 + 8 * hi];
                const __bf16* bl = &ldsW[1 * 4096 + cl * 32 + 8 * hi];
                bhi.h[0] = *(const v8bf*)bh;  bhi.h[1] = *(const v8bf*)(bh + 16);
                blo.h[0] = *(const v8bf*)bl;  blo.h[1] = *(const v8bf*)(bl + 16);
                acc[nt] = __builtin_amdgcn_wmma_f32_16x16x32_bf16(
                    false, ahi.v, false, bhi.v, (short)0, acc[nt], false, false);
                acc[nt] = __builtin_amdgcn_wmma_f32_16x16x32_bf16(
                    false, ahi.v, false, blo.v, (short)0, acc[nt], false, false);
                acc[nt] = __builtin_amdgcn_wmma_f32_16x16x32_bf16(
                    false, alo.v, false, bhi.v, (short)0, acc[nt], false, false);
            }
        }
    }
    if (active) {
        #pragma unroll
        for (int nt = 0; nt < NT_W; ++nt) {
            int col = n0 + nt * 16 + l16;
            float bv = bias[col];
            #pragma unroll
            for (int r = 0; r < 8; ++r) {
                int orow = mtile * 16 + r + 8 * hi;   // C/D layout: M = r + 8*hi
                if (orow < Nrows) out[(size_t)orow * FD + col] = acc[nt][r] + bv;
            }
        }
    }
}

// ---------------- per-node init -------------------------------------------
__global__ void init_nodes_kernel(float* __restrict__ m, float* __restrict__ den, long long n)
{
    long long i = (long long)blockIdx.x * 256 + threadIdx.x;
    if (i >= n) return;
    m[i]   = -__builtin_inff();
    den[i] = 0.0f;
}

// monotonic float atomic-max via int/uint atomics
__device__ __forceinline__ void atomicMaxF(float* addr, float v)
{
    if (v >= 0.0f) atomicMax((int*)addr, __float_as_int(v));
    else           atomicMin((unsigned int*)addr, __float_as_uint(v));
}

// ---------------- edge logits: alpha = (q[dst].k[src])/16, segment max ------
__global__ void edge_logits_kernel(const float* __restrict__ q, const float* __restrict__ k,
                                   const long long* __restrict__ ei, long long E,
                                   float* __restrict__ alpha, float* __restrict__ mbuf)
{
    long long e = (long long)blockIdx.x * 8 + (threadIdx.x >> 5);
    if (e >= E) return;
    int lane = threadIdx.x & 31;
    long long s = ei[e];
    long long d = ei[E + e];
    const float4* qd = (const float4*)(q + (size_t)d * FD);
    const float4* ks = (const float4*)(k + (size_t)s * FD);
    float4 a0 = qd[lane * 2], a1 = qd[lane * 2 + 1];
    float4 b0 = ks[lane * 2], b1 = ks[lane * 2 + 1];
    float sum = a0.x * b0.x + a0.y * b0.y + a0.z * b0.z + a0.w * b0.w
              + a1.x * b1.x + a1.y * b1.y + a1.z * b1.z + a1.w * b1.w;
    #pragma unroll
    for (int off = 16; off > 0; off >>= 1) sum += __shfl_xor(sum, off, 32);
    if (lane == 0) {
        float al = sum * 0.0625f;       // 1/sqrt(256)
        alpha[e] = al;
        atomicMaxF(mbuf + d, al);
    }
}

// ---------------- exp + segment sum ----------------------------------------
__global__ void edge_exp_kernel(const long long* __restrict__ ei, long long E,
                                float* __restrict__ alpha, const float* __restrict__ mbuf,
                                float* __restrict__ den)
{
    long long e = (long long)blockIdx.x * 256 + threadIdx.x;
    if (e >= E) return;
    long long d  = ei[E + e];
    float ee = __expf(alpha[e] - mbuf[d]);   // mbuf[d] finite: this edge exists
    alpha[e] = ee;
    atomicAdd(den + d, ee);
}

// ---------------- aggregation: out[dst] += attn * v[src] --------------------
__global__ void edge_agg_kernel(const float* __restrict__ v, const long long* __restrict__ ei,
                                long long E, const float* __restrict__ alpha,
                                const float* __restrict__ den, float* __restrict__ out)
{
    long long e = (long long)blockIdx.x * 8 + (threadIdx.x >> 5);
    if (e >= E) return;
    int lane = threadIdx.x & 31;
    long long s = ei[e];
    long long d = ei[E + e];
    float attn = alpha[e] / (den[d] + 1e-16f);
    const float4* vs = (const float4*)(v + (size_t)s * FD);
    float4 x0 = vs[lane * 2], x1 = vs[lane * 2 + 1];
    float* ob = out + (size_t)d * FD + lane * 8;
    atomicAdd(ob + 0, attn * x0.x);
    atomicAdd(ob + 1, attn * x0.y);
    atomicAdd(ob + 2, attn * x0.z);
    atomicAdd(ob + 3, attn * x0.w);
    atomicAdd(ob + 4, attn * x1.x);
    atomicAdd(ob + 5, attn * x1.y);
    atomicAdd(ob + 6, attn * x1.z);
    atomicAdd(ob + 7, attn * x1.w);
}

// ---------------- relu ------------------------------------------------------
__global__ void relu_kernel(float* __restrict__ h, long long n)
{
    long long i = (long long)blockIdx.x * 256 + threadIdx.x;
    if (i < n) h[i] = fmaxf(h[i], 0.0f);
}

// ---------------------------------------------------------------------------
extern "C" void kernel_launch(void* const* d_in, const int* in_sizes, int n_in,
                              void* d_out, int out_size, void* d_ws, size_t ws_size,
                              hipStream_t stream)
{
    (void)n_in; (void)out_size; (void)ws_size;
    const float*     x   = (const float*)d_in[0];
    const long long* ei  = (const long long*)d_in[1];   // int64 edge_index [2,E]
    const float* Wq = (const float*)d_in[2];
    const float* bq = (const float*)d_in[3];
    const float* Wk = (const float*)d_in[4];
    const float* bk = (const float*)d_in[5];
    const float* Wv = (const float*)d_in[6];
    const float* bv = (const float*)d_in[7];
    const float* Ws = (const float*)d_in[8];
    const float* bs = (const float*)d_in[9];
    float* out = (float*)d_out;

    const long long Nn = (long long)in_sizes[0] / FD;
    const long long E  = (long long)in_sizes[1] / 2;
    const int       L  = in_sizes[2] / (FD * FD);

    // ---- workspace carving (256B-aligned) ----
    char* ws = (char*)d_ws;
    size_t off = 0;
    auto carve = [&](size_t bytes) -> char* {
        char* p = ws + off;
        off += (bytes + 255) & ~(size_t)255;
        return p;
    };
    float*  qb    = (float*) carve((size_t)Nn * FD * 4);
    float*  kb2   = (float*) carve((size_t)Nn * FD * 4);
    float*  vb    = (float*) carve((size_t)Nn * FD * 4);
    float*  htmp  = (float*) carve((size_t)Nn * FD * 4);
    __bf16* xhi   = (__bf16*)carve((size_t)Nn * FD * 2);
    __bf16* xlo   = (__bf16*)carve((size_t)Nn * FD * 2);
    __bf16* Whi   = (__bf16*)carve((size_t)L * 4 * FD * FD * 2);
    __bf16* Wlo   = (__bf16*)carve((size_t)L * 4 * FD * FD * 2);
    float*  alpha = (float*) carve((size_t)E * 4);
    float*  mb    = (float*) carve((size_t)Nn * 4);
    float*  den   = (float*) carve((size_t)Nn * 4);

    // one-time (per call) weight transpose + bf16 split
    {
        long long tot = (long long)L * 4 * FD * FD;
        prep_w_kernel<<<(unsigned)((tot + 255) / 256), 256, 0, stream>>>(
            Wq, Wk, Wv, Ws, Whi, Wlo, L);
    }

    const int  mt_cnt = (int)((Nn + 15) / 16);
    const dim3 ggrid((unsigned)((mt_cnt + 7) / 8), 2);
    const long long nd = Nn * FD;

    for (int l = 0; l < L; ++l) {
        const float* hin  = (l == 0)     ? x   : htmp;
        float*       hout = (l == L - 1) ? out : htmp;

        cvt_kernel<<<(unsigned)((nd + 255) / 256), 256, 0, stream>>>(hin, xhi, xlo, nd);

        const float* biases[4] = {bq, bk, bv, bs};
        float*       outs[4]   = {qb, kb2, vb, hout};
        for (int w = 0; w < 4; ++w) {
            size_t woff = ((size_t)l * 4 + w) * FD * FD;
            gemm_bf16x3_kernel<<<ggrid, 256, 0, stream>>>(
                xhi, xlo, Whi + woff, Wlo + woff,
                biases[w] + (size_t)l * FD, outs[w], (int)Nn);
        }

        init_nodes_kernel<<<(unsigned)((Nn + 255) / 256), 256, 0, stream>>>(mb, den, Nn);
        edge_logits_kernel<<<(unsigned)((E + 7) / 8), 256, 0, stream>>>(qb, kb2, ei, E, alpha, mb);
        edge_exp_kernel<<<(unsigned)((E + 255) / 256), 256, 0, stream>>>(ei, E, alpha, mb, den);
        edge_agg_kernel<<<(unsigned)((E + 7) / 8), 256, 0, stream>>>(vb, ei, E, alpha, den, hout);

        if (l < L - 1)
            relu_kernel<<<(unsigned)((nd + 255) / 256), 256, 0, stream>>>(hout, nd);
    }
}
